// VectorQuantizer_65575560675404
// MI455X (gfx1250) — matrix operand order repaired and checked by reading, not compiled, not run on hardware
//
#include <hip/hip_runtime.h>
#include <math.h>

// ---- types (trivial ext-vector types to keep unions trivially constructible) ----
typedef __attribute__((ext_vector_type(16))) __bf16        v16bf;
typedef __attribute__((ext_vector_type(8)))  float         v8f;
typedef __attribute__((ext_vector_type(4)))  unsigned int  u32x4;
typedef __attribute__((ext_vector_type(2)))  unsigned int  u32x2;
typedef __attribute__((ext_vector_type(4)))  float         f32x4;

#define N_VEC   8192           // B*K = 32*256 input vectors
#define N_CODE  8192           // codebook entries
#define DIM     256
#define NELEM   (N_VEC * DIM)  // 2097152

union Frag { v16bf v; u32x4 q[2]; };

__device__ __forceinline__ unsigned short f2bf(float f) {
  unsigned int u = __float_as_uint(f);
  u += 0x7FFFu + ((u >> 16) & 1u);      // round-to-nearest-even
  return (unsigned short)(u >> 16);
}

// ---------------------------------------------------------------------------
// Kernel A: fp32 -> bf16 conversion of inputs + codebook, zero histogram
// ---------------------------------------------------------------------------
__global__ void vq_prep(const f32x4* __restrict__ inp4, const f32x4* __restrict__ cb4,
                        u32x2* __restrict__ inp_bf4, u32x2* __restrict__ cb_bf4,
                        unsigned int* __restrict__ counts) {
  int i = blockIdx.x * blockDim.x + threadIdx.x;   // 524288 threads, 4 elems each
  f32x4 a = inp4[i];
  f32x4 b = cb4[i];
  u32x2 oa, ob;
  oa.x = (unsigned int)f2bf(a.x) | ((unsigned int)f2bf(a.y) << 16);
  oa.y = (unsigned int)f2bf(a.z) | ((unsigned int)f2bf(a.w) << 16);
  ob.x = (unsigned int)f2bf(b.x) | ((unsigned int)f2bf(b.y) << 16);
  ob.y = (unsigned int)f2bf(b.z) | ((unsigned int)f2bf(b.w) << 16);
  inp_bf4[i] = oa;
  cb_bf4[i]  = ob;
  if (i < N_CODE) counts[i] = 0u;
}

// ---------------------------------------------------------------------------
// Kernel B: halfnorm[j] = 0.5 * ||codebook[j]||^2   (one wave per row)
// ---------------------------------------------------------------------------
__global__ void vq_halfnorm(const float* __restrict__ cb, float* __restrict__ hn) {
  int lane = threadIdx.x & 31;
  int row  = blockIdx.x * (blockDim.x >> 5) + (threadIdx.x >> 5);
  const float* r = cb + row * DIM;
  float s = 0.f;
  #pragma unroll
  for (int j = 0; j < DIM / 32; ++j) { float v = r[lane + 32 * j]; s += v * v; }
  #pragma unroll
  for (int m = 16; m >= 1; m >>= 1) s += __shfl_xor(s, m, 32);
  if (lane == 0) hn[row] = 0.5f * s;
}

// ---------------------------------------------------------------------------
// Kernel C: WMMA bf16 GEMM + running argmax of (x.e - 0.5||e||^2)
// 2048 waves: 512 row-tiles x 4 column splits; B tiles ping-pong double
// buffered so each wave keeps 16 b128 loads in flight behind its WMMA chain.
// ---------------------------------------------------------------------------
__global__ void __launch_bounds__(256) vq_argmax_wmma(
    const unsigned short* __restrict__ inp_bf,
    const unsigned short* __restrict__ cb_bf,
    const float* __restrict__ hn,
    float* __restrict__ pScore, int* __restrict__ pIdx) {
  const int lane    = threadIdx.x & 31;
  const int gwid    = blockIdx.x * 8 + (threadIdx.x >> 5);
  const int rowTile = gwid >> 2;        // 0..511
  const int split   = gwid & 3;         // 0..3
  const int half    = lane >> 4;        // 0/1
  const int l16     = lane & 15;

  // A fragments for this 16-row tile, all of K=256 (8 chunks of 32), in VGPRs
  Frag a[8];
  {
    const unsigned short* arow = inp_bf + (rowTile * 16 + l16) * DIM + half * 8;
    #pragma unroll
    for (int k = 0; k < 8; ++k) {
      a[k].q[0] = *(const u32x4*)(arow + k * 32);
      a[k].q[1] = *(const u32x4*)(arow + k * 32 + 16);
    }
  }

  float best[8];
  int   bidx[8];
  #pragma unroll
  for (int r = 0; r < 8; ++r) { best[r] = -3.4e38f; bidx[r] = 0; }

  const int ct0 = split * 128;          // 128 column tiles per split
  const unsigned short* bbase = cb_bf + l16 * DIM + half * 8;

  auto loadB = [&](Frag* b, int tile) {
    const unsigned short* brow = bbase + tile * (16 * DIM);
    #pragma unroll
    for (int k = 0; k < 8; ++k) {
      b[k].q[0] = *(const u32x4*)(brow + k * 32);
      b[k].q[1] = *(const u32x4*)(brow + k * 32 + 16);
    }
  };
  auto process = [&](const Frag* b, int c) {
    v8f acc = {};
    #pragma unroll
    for (int k = 0; k < 8; ++k)
      acc = __builtin_amdgcn_wmma_f32_16x16x32_bf16(
          false, a[k].v, false, b[k].v, (short)0, acc, false, false);
    const int   col = (ct0 + c) * 16 + l16;
    const float nb  = -hn[col];
    #pragma unroll
    for (int r = 0; r < 8; ++r) {
      float s = acc[r] + nb;
      if (s > best[r]) { best[r] = s; bidx[r] = col; }
    }
  };

  Frag bufA[8], bufB[8];
  loadB(bufA, ct0);
  for (int c = 0; c < 128; c += 2) {
    // prefetch a few tiles ahead into the cache hierarchy (global_prefetch_b8)
    __builtin_prefetch(bbase + (ct0 + ((c + 4 < 128) ? c + 4 : 127)) * (16 * DIM), 0, 3);
    loadB(bufB, ct0 + c + 1);
    process(bufA, c);
    loadB(bufA, ct0 + ((c + 2 < 128) ? c + 2 : 127));
    process(bufB, c + 1);
  }

  // reduce across the 16 lanes holding the same row (xor masks stay in-half)
  #pragma unroll
  for (int r = 0; r < 8; ++r) {
    float s = best[r]; int i = bidx[r];
    #pragma unroll
    for (int m = 8; m >= 1; m >>= 1) {
      float s2 = __shfl_xor(s, m, 32);
      int   i2 = __shfl_xor(i, m, 32);
      if (s2 > s || (s2 == s && i2 < i)) { s = s2; i = i2; }
    }
    if (l16 == 0) {
      int row = rowTile * 16 + r + half * 8;
      pScore[split * N_VEC + row] = s;
      pIdx  [split * N_VEC + row] = i;
    }
  }
}

// ---------------------------------------------------------------------------
// Kernel C2: merge 4 column-split partials per row
// ---------------------------------------------------------------------------
__global__ void vq_combine(const float* __restrict__ pScore, const int* __restrict__ pIdx,
                           int* __restrict__ bestIdx) {
  int n = blockIdx.x * blockDim.x + threadIdx.x;    // 8192
  float bs = pScore[n]; int bi = pIdx[n];
  #pragma unroll
  for (int s = 1; s < 4; ++s) {
    float sc = pScore[s * N_VEC + n]; int id = pIdx[s * N_VEC + n];
    if (sc > bs || (sc == bs && id < bi)) { bs = sc; bi = id; }
  }
  bestIdx[n] = bi;
}

// ---------------------------------------------------------------------------
// Kernel D: gather quantized rows (STE form), per-row loss partial, histogram
// ---------------------------------------------------------------------------
__global__ void vq_gather(const float* __restrict__ inp, const float* __restrict__ cb,
                          const int* __restrict__ bestIdx, float* __restrict__ outq,
                          float* __restrict__ lossPartial, unsigned int* __restrict__ counts) {
  __shared__ float sdata[256];
  const int row = blockIdx.x;
  const int t   = threadIdx.x;
  const int idx = bestIdx[row];
  float q = cb[idx * DIM + t];
  float x = inp[row * DIM + t];
  outq[row * DIM + t] = x + (q - x);     // mirror reference STE rounding
  float d = q - x;
  sdata[t] = d * d;
  __syncthreads();
  #pragma unroll
  for (int s = 128; s >= 1; s >>= 1) {
    if (t < s) sdata[t] += sdata[t + s];
    __syncthreads();
  }
  if (t == 0) {
    lossPartial[row] = sdata[0];
    atomicAdd(&counts[idx], 1u);
  }
}

// ---------------------------------------------------------------------------
// Kernel E: loss scalar + perplexity (fixed-order reductions -> deterministic)
// ---------------------------------------------------------------------------
__global__ void vq_finalize(const unsigned int* __restrict__ counts,
                            const float* __restrict__ lossPartial,
                            float* __restrict__ out) {
  __shared__ float se[256], sl[256];
  const int t = threadIdx.x;
  float e = 0.f, ls = 0.f;
  for (int j = t; j < N_CODE; j += 256) {
    float p = (float)counts[j] * (1.0f / (float)N_VEC);
    e  += p * logf(p + 1e-10f);
    ls += lossPartial[j];
  }
  se[t] = e; sl[t] = ls;
  __syncthreads();
  #pragma unroll
  for (int s = 128; s >= 1; s >>= 1) {
    if (t < s) { se[t] += se[t + s]; sl[t] += sl[t + s]; }
    __syncthreads();
  }
  if (t == 0) {
    out[NELEM]     = 1.25f * sl[0] / (float)NELEM;  // q_latent + 0.25 * e_latent
    out[NELEM + 1] = expf(-se[0]);                  // perplexity
  }
}

// ---------------------------------------------------------------------------
extern "C" void kernel_launch(void* const* d_in, const int* in_sizes, int n_in,
                              void* d_out, int out_size, void* d_ws, size_t ws_size,
                              hipStream_t stream) {
  const float* inp = (const float*)d_in[0];   // [32,256,256] fp32
  const float* cb  = (const float*)d_in[1];   // [8192,256]  fp32
  float* out = (float*)d_out;                 // 2097152 + loss + perplexity

  char* ws = (char*)d_ws;
  unsigned short* inp_bf = (unsigned short*)(ws);                 // 4 MB
  unsigned short* cb_bf  = (unsigned short*)(ws + 4194304);       // 4 MB
  float*        hn      = (float*)(ws + 8388608);                 // 32 KB
  float*        pS      = (float*)(ws + 8421376);                 // 128 KB
  int*          pI      = (int*)  (ws + 8552448);                 // 128 KB
  int*          bIdx    = (int*)  (ws + 8683520);                 // 32 KB
  unsigned int* counts  = (unsigned int*)(ws + 8716288);          // 32 KB
  float*        lossP   = (float*)(ws + 8749056);                 // 32 KB

  vq_prep       <<<2048, 256, 0, stream>>>((const f32x4*)inp, (const f32x4*)cb,
                                           (u32x2*)inp_bf, (u32x2*)cb_bf, counts);
  vq_halfnorm   <<<1024, 256, 0, stream>>>(cb, hn);
  vq_argmax_wmma<<<256,  256, 0, stream>>>(inp_bf, cb_bf, hn, pS, pI);
  vq_combine    <<<32,   256, 0, stream>>>(pS, pI, bIdx);
  vq_gather     <<<8192, 256, 0, stream>>>(inp, cb, bIdx, out, lossP, counts);
  vq_finalize   <<<1,    256, 0, stream>>>(counts, lossP, out);
}
